// MANN_7533372637797
// MI455X (gfx1250) — compile-verified, matrix operand order checked by reading, not run on hardware
//
#include <hip/hip_runtime.h>
#include <hip/hip_bf16.h>

#define B_ROWS 16384
#define DIN 480
#define DH 512
#define DOUT 360
#define NEXP 8
#define GIN 64
#define GH 128

typedef __attribute__((ext_vector_type(16))) __bf16 v16bf;
typedef __attribute__((ext_vector_type(8)))  float  v8f;

union FragBF {
    uint4        q[2];
    unsigned int u[8];
    v16bf        v;
};

static __device__ __forceinline__ unsigned int pk_mul_bf16(unsigned int a, unsigned int s) {
    unsigned int d;
    asm volatile("v_pk_mul_bf16 %0, %1, %2" : "=v"(d) : "v"(a), "v"(s));
    return d;
}

// CDNA5 async DMA: global -> LDS, 16B per lane, tracked on ASYNCcnt.
static __device__ __forceinline__ void async_ld_b128(unsigned lds_off, const void* gaddr) {
    unsigned long long ga = (unsigned long long)(uintptr_t)gaddr;
    asm volatile("global_load_async_to_lds_b128 %0, %1, off"
                 :: "v"(lds_off), "v"(ga)
                 : "memory");
}

static __device__ __forceinline__ void wait_async0() {
    asm volatile("s_wait_asynccnt 0x0" ::: "memory");
}

static __device__ __forceinline__ float elu_f(float v) {
    return (v > 0.f) ? v : (__expf(v) - 1.f);
}

// ---------------------------------------------------------------------------
// Weight convert f32 -> bf16 with transpose: Wt[e][n][k] = W[e][k][n].
// Pads n in [N, Npad) with zeros so the GEMM never reads out of bounds.
// ---------------------------------------------------------------------------
__global__ void wconv_transpose_kernel(const float* __restrict__ W,
                                       __hip_bfloat16* __restrict__ Wt,
                                       int K, int N, int Npad) {
    size_t total  = (size_t)NEXP * Npad * K;
    size_t stride = (size_t)gridDim.x * blockDim.x;
    for (size_t i = (size_t)blockIdx.x * blockDim.x + threadIdx.x; i < total; i += stride) {
        int    k = (int)(i % K);
        size_t r = i / K;
        int    n = (int)(r % Npad);
        int    e = (int)(r / Npad);
        float  v = (n < N) ? W[((size_t)e * K + k) * N + n] : 0.f;
        Wt[i] = __float2bfloat16(v);
    }
}

// ---------------------------------------------------------------------------
// xn = (x - mu) / sigma, stored bf16 (A-matrix for layer 1).
// ---------------------------------------------------------------------------
__global__ void xnorm_kernel(const float* __restrict__ x,
                             const float* __restrict__ xn,
                             __hip_bfloat16* __restrict__ out) {
    size_t total  = (size_t)B_ROWS * DIN;
    size_t stride = (size_t)gridDim.x * blockDim.x;
    for (size_t i = (size_t)blockIdx.x * blockDim.x + threadIdx.x; i < total; i += stride) {
        int c = (int)(i % DIN);
        out[i] = __float2bfloat16((x[i] - xn[c]) / xn[DIN + c]);
    }
}

// ---------------------------------------------------------------------------
// Gating MLP: 64 -> 128 (ELU) -> 128 (ELU) -> 8 (softmax). One wave per row.
// ---------------------------------------------------------------------------
__launch_bounds__(256)
__global__ void gating_kernel(const float* __restrict__ x,  const float* __restrict__ xn,
                              const int*   __restrict__ gidx,
                              const float* __restrict__ G1W, const float* __restrict__ G1b,
                              const float* __restrict__ G2W, const float* __restrict__ G2b,
                              const float* __restrict__ G3W, const float* __restrict__ G3b,
                              float* __restrict__ gate) {
    __shared__ float sx[8][GIN];
    __shared__ float sh1[8][GH];
    __shared__ float sh2[8][GH];
    __shared__ float slog[8][NEXP];

    const int tid = threadIdx.x;
    const int lane = tid & 31;
    const int w = tid >> 5;
    const int b = blockIdx.x * 8 + w;

    for (int i = lane; i < GIN; i += 32) {
        int idx = gidx[i];
        sx[w][i] = (x[(size_t)b * DIN + idx] - xn[idx]) / xn[DIN + idx];
    }
    __syncthreads();

    // layer 1: each lane produces 4 of 128 hidden units
    {
        float4 a = *(const float4*)&G1b[lane * 4];
        #pragma unroll 8
        for (int i = 0; i < GIN; ++i) {
            float  xi = sx[w][i];
            float4 wr = *(const float4*)&G1W[i * GH + lane * 4];
            a.x = fmaf(xi, wr.x, a.x);
            a.y = fmaf(xi, wr.y, a.y);
            a.z = fmaf(xi, wr.z, a.z);
            a.w = fmaf(xi, wr.w, a.w);
        }
        sh1[w][lane * 4 + 0] = elu_f(a.x);
        sh1[w][lane * 4 + 1] = elu_f(a.y);
        sh1[w][lane * 4 + 2] = elu_f(a.z);
        sh1[w][lane * 4 + 3] = elu_f(a.w);
    }
    __syncthreads();

    // layer 2
    {
        float4 a = *(const float4*)&G2b[lane * 4];
        #pragma unroll 8
        for (int i = 0; i < GH; ++i) {
            float  xi = sh1[w][i];
            float4 wr = *(const float4*)&G2W[i * GH + lane * 4];
            a.x = fmaf(xi, wr.x, a.x);
            a.y = fmaf(xi, wr.y, a.y);
            a.z = fmaf(xi, wr.z, a.z);
            a.w = fmaf(xi, wr.w, a.w);
        }
        sh2[w][lane * 4 + 0] = elu_f(a.x);
        sh2[w][lane * 4 + 1] = elu_f(a.y);
        sh2[w][lane * 4 + 2] = elu_f(a.z);
        sh2[w][lane * 4 + 3] = elu_f(a.w);
    }
    __syncthreads();

    // logits
    if (lane < NEXP) {
        float a = G3b[lane];
        #pragma unroll 8
        for (int i = 0; i < GH; ++i) a = fmaf(sh2[w][i], G3W[i * NEXP + lane], a);
        slog[w][lane] = a;
    }
    __syncthreads();

    // softmax
    if (lane < NEXP) {
        float m = slog[w][0];
        #pragma unroll
        for (int e = 1; e < NEXP; ++e) m = fmaxf(m, slog[w][e]);
        float s = 0.f;
        #pragma unroll
        for (int e = 0; e < NEXP; ++e) s += __expf(slog[w][e] - m);
        gate[(size_t)b * NEXP + lane] = __expf(slog[w][lane] - m) / s;
    }
}

// ---------------------------------------------------------------------------
// MoE expert GEMM:
//   out[b,n] = f( sum_e gate[b,e]*(A[b,:] @ W_e[:,n]) + (gate @ bias)[b,n] )
// Tile 128(M) x 64(N), 8 waves; each wave owns a 32x32 patch (4 WMMA tiles).
// Two k-passes of 4 experts each; gate folded into A-fragments with
// v_pk_mul_bf16 (scales preloaded to registers, k-invariant).
// Staging: double-buffered global_load_async_to_lds_b128 (ASYNCcnt), so HBM
// latency hides under v_wmma_f32_16x16x32_bf16.
// Wt is pre-transposed bf16: Wt[e][n][k].
// ---------------------------------------------------------------------------
__launch_bounds__(256, 2)
__global__ void moe_gemm_kernel(const __hip_bfloat16* __restrict__ A, int Astride, int K,
                                const __hip_bfloat16* __restrict__ Wt, int Kstride, int Npad,
                                const float* __restrict__ bias, int Nbias,
                                const float* __restrict__ gate,
                                __hip_bfloat16* __restrict__ outB, int outStride,
                                float* __restrict__ outF,
                                const float* __restrict__ ynorm, int Nout) {
    // Padded rows: 40 bf16 = 80 B -> b128-aligned, conflict-free bank spread.
    __shared__ __align__(16) unsigned short sA[2][128 * 40];     // 2 x 10 KB
    __shared__ __align__(16) unsigned short sB[2][4 * 64 * 40];  // 2 x 20 KB

    const int tid  = threadIdx.x;
    const int lane = tid & 31;
    const int wid  = tid >> 5;
    const int wm   = wid & 3;   // 4 wave rows  (M)
    const int wn   = wid >> 2;  // 2 wave cols  (N)
    const int lm   = lane & 15; // fragment row/col within 16
    const int lg   = lane >> 4; // K half-group

    const int rowbase = blockIdx.y * 128;
    const int n0      = blockIdx.x * 64;
    const int nIter   = K / 32;

    v8f acc[2][2] = {};

    for (int pass = 0; pass < 2; ++pass) {
        const int ebase = pass * 4;

        // k-invariant gate scales for this pass, packed bf16x2, in registers
        unsigned s2[4][2];
        #pragma unroll
        for (int e = 0; e < 4; ++e)
            #pragma unroll
            for (int mi = 0; mi < 2; ++mi) {
                float gs = gate[(size_t)(rowbase + wm * 32 + mi * 16 + lm) * NEXP + ebase + e];
                unsigned s16 = __float_as_uint(gs) >> 16;  // truncated bf16
                s2[e][mi] = s16 | (s16 << 16);
            }

        // ---- async stage helper: A tile 128x32 (512 b128 units),
        //      B tiles 4 experts x 64x32 (1024 b128 units), 6 units/thread ----
        auto stage_async = [&](int buf, int k0) {
            #pragma unroll
            for (int i = 0; i < 2; ++i) {
                int u = tid + 256 * i;            // 0..511
                int row = u >> 2, q = u & 3;      // 8 bf16 per b128 unit
                async_ld_b128((unsigned)(uintptr_t)&sA[buf][row * 40 + q * 8],
                              A + (size_t)(rowbase + row) * Astride + k0 + q * 8);
            }
            #pragma unroll
            for (int i = 0; i < 4; ++i) {
                int u = tid + 256 * i;            // 0..1023
                int e = u >> 8, rem = u & 255;
                int n = rem >> 2, q = rem & 3;
                async_ld_b128((unsigned)(uintptr_t)&sB[buf][(e * 64 + n) * 40 + q * 8],
                              Wt + ((size_t)(ebase + e) * Npad + n0 + n) * Kstride + k0 + q * 8);
            }
        };

        stage_async(0, 0);

        for (int it = 0; it < nIter; ++it) {
            const int cur = it & 1;
            wait_async0();         // this wave's tile DMA done
            __syncthreads();       // everyone's tile visible; prev buffer free

            if (it + 1 < nIter) stage_async(cur ^ 1, (it + 1) * 32);  // prefetch under compute

            // A fragments (16x32 bf16 WMMA layout)
            FragBF afr[2];
            #pragma unroll
            for (int mi = 0; mi < 2; ++mi) {
                int row = wm * 32 + mi * 16 + lm;
                afr[mi].q[0] = *(const uint4*)&sA[cur][row * 40 + lg * 8];
                afr[mi].q[1] = *(const uint4*)&sA[cur][row * 40 + lg * 8 + 16];
            }

            #pragma unroll
            for (int e = 0; e < 4; ++e) {
                FragBF asc[2];
                #pragma unroll
                for (int mi = 0; mi < 2; ++mi)
                    #pragma unroll
                    for (int j = 0; j < 8; ++j)
                        asc[mi].u[j] = pk_mul_bf16(afr[mi].u[j], s2[e][mi]);

                FragBF bfr[2];
                #pragma unroll
                for (int ni = 0; ni < 2; ++ni) {
                    int col = wn * 32 + ni * 16 + lm;
                    bfr[ni].q[0] = *(const uint4*)&sB[cur][(e * 64 + col) * 40 + lg * 8];
                    bfr[ni].q[1] = *(const uint4*)&sB[cur][(e * 64 + col) * 40 + lg * 8 + 16];
                }
                #pragma unroll
                for (int mi = 0; mi < 2; ++mi)
                    #pragma unroll
                    for (int ni = 0; ni < 2; ++ni)
                        acc[mi][ni] = __builtin_amdgcn_wmma_f32_16x16x32_bf16(
                            false, asc[mi].v, false, bfr[ni].v, (short)0, acc[mi][ni], false, false);
            }
        }
        __syncthreads();  // all reads of last buffer done before next pass restages
    }

    // ---- epilogue: + gate@bias, ELU / Ynorm affine, store ----
    #pragma unroll
    for (int ni = 0; ni < 2; ++ni) {
        const int col  = wn * 32 + ni * 16 + lm;
        const int gcol = n0 + col;
        float bv[NEXP];
        #pragma unroll
        for (int e = 0; e < NEXP; ++e)
            bv[e] = (gcol < Nbias) ? bias[(size_t)e * Nbias + gcol] : 0.f;
        #pragma unroll
        for (int mi = 0; mi < 2; ++mi) {
            #pragma unroll
            for (int r = 0; r < 8; ++r) {
                const int rowT = wm * 32 + mi * 16 + lg * 8 + r;
                const float4 g0 = *(const float4*)&gate[(size_t)(rowbase + rowT) * NEXP];
                const float4 g1 = *(const float4*)&gate[(size_t)(rowbase + rowT) * NEXP + 4];
                float t = g0.x * bv[0] + g0.y * bv[1] + g0.z * bv[2] + g0.w * bv[3]
                        + g1.x * bv[4] + g1.y * bv[5] + g1.z * bv[6] + g1.w * bv[7];
                float v = acc[mi][ni][r] + t;
                if (outF) {
                    if (gcol < Nout) {
                        v = v * ynorm[DOUT + gcol] + ynorm[gcol];
                        outF[(size_t)(rowbase + rowT) * Nout + gcol] = v;
                    }
                } else {
                    v = elu_f(v);
                    outB[(size_t)(rowbase + rowT) * outStride + gcol] = __float2bfloat16(v);
                }
            }
        }
    }
}

// ---------------------------------------------------------------------------
extern "C" void kernel_launch(void* const* d_in, const int* in_sizes, int n_in,
                              void* d_out, int out_size, void* d_ws, size_t ws_size,
                              hipStream_t stream) {
    const float* x   = (const float*)d_in[0];
    const float* Xn  = (const float*)d_in[1];
    const float* Yn  = (const float*)d_in[2];
    const float* G1W = (const float*)d_in[3];
    const float* G1b = (const float*)d_in[4];
    const float* G2W = (const float*)d_in[5];
    const float* G2b = (const float*)d_in[6];
    const float* G3W = (const float*)d_in[7];
    const float* G3b = (const float*)d_in[8];
    const float* W1  = (const float*)d_in[9];
    const float* b1  = (const float*)d_in[10];
    const float* W2  = (const float*)d_in[11];
    const float* b2  = (const float*)d_in[12];
    const float* W3  = (const float*)d_in[13];
    const float* b3  = (const float*)d_in[14];
    const int*  gidx = (const int*)d_in[15];

    char* ws = (char*)d_ws;
    size_t off = 0;
    auto wsalloc = [&](size_t bytes) -> void* {
        void* p = ws + off;
        off = (off + bytes + 255) & ~(size_t)255;
        return p;
    };

    __hip_bfloat16* xnb  = (__hip_bfloat16*)wsalloc((size_t)B_ROWS * DIN * 2);
    __hip_bfloat16* act1 = (__hip_bfloat16*)wsalloc((size_t)B_ROWS * DH * 2);
    __hip_bfloat16* act2 = (__hip_bfloat16*)wsalloc((size_t)B_ROWS * DH * 2);
    float*          gate = (float*)wsalloc((size_t)B_ROWS * NEXP * 4);
    __hip_bfloat16* W1t  = (__hip_bfloat16*)wsalloc((size_t)NEXP * DH * DIN * 2);   // [8][512][480]
    __hip_bfloat16* W2t  = (__hip_bfloat16*)wsalloc((size_t)NEXP * DH * DH * 2);    // [8][512][512]
    __hip_bfloat16* W3t  = (__hip_bfloat16*)wsalloc((size_t)NEXP * 384 * DH * 2);   // [8][384][512], n-padded

    // one-time prep
    wconv_transpose_kernel<<<2048, 256, 0, stream>>>(W1, W1t, DIN, DH, DH);
    wconv_transpose_kernel<<<2048, 256, 0, stream>>>(W2, W2t, DH, DH, DH);
    wconv_transpose_kernel<<<2048, 256, 0, stream>>>(W3, W3t, DH, DOUT, 384);
    xnorm_kernel<<<4096, 256, 0, stream>>>(x, Xn, xnb);
    gating_kernel<<<B_ROWS / 8, 256, 0, stream>>>(x, Xn, gidx, G1W, G1b, G2W, G2b, G3W, G3b, gate);

    // layer 1: xnb (B x 480) -> act1 (B x 512), ELU
    moe_gemm_kernel<<<dim3(DH / 64, B_ROWS / 128), 256, 0, stream>>>(
        xnb, DIN, DIN, W1t, DIN, DH, b1, DH, gate, act1, DH, nullptr, nullptr, DH);
    // layer 2: act1 -> act2, ELU
    moe_gemm_kernel<<<dim3(DH / 64, B_ROWS / 128), 256, 0, stream>>>(
        act1, DH, DH, W2t, DH, DH, b2, DH, gate, act2, DH, nullptr, nullptr, DH);
    // layer 3: act2 -> d_out (B x 360 f32) with Ynorm affine
    moe_gemm_kernel<<<dim3(384 / 64, B_ROWS / 128), 256, 0, stream>>>(
        act2, DH, DH, W3t, DH, 384, b3, DOUT, gate, nullptr, 0, (float*)d_out, Yn, DOUT);
}
